// MPNEncoder_24240795418898
// MI455X (gfx1250) — compile-verified, matrix operand order checked by reading, not run on hardware
//
#include <hip/hip_runtime.h>
#include <hip/hip_bf16.h>

typedef __bf16 bf16x4  __attribute__((ext_vector_type(4)));
typedef __bf16 bf16x8  __attribute__((ext_vector_type(8)));
typedef __bf16 bf16x16 __attribute__((ext_vector_type(16)));
typedef float  floatx4 __attribute__((ext_vector_type(4)));
typedef float  floatx8 __attribute__((ext_vector_type(8)));
typedef unsigned int u32x4 __attribute__((ext_vector_type(4)));
typedef int    i32x4   __attribute__((ext_vector_type(4)));
typedef int    i32x8   __attribute__((ext_vector_type(8)));

#define HIDDEN 256
#define N_MOLS 5000
#define ATOMS_PER_MOL 30
#define N_ATOMS (N_MOLS * ATOMS_PER_MOL)   // 150000 (+1 pad row)
#define N_BONDS 320000                      // (+1 pad row)
#define MAX_NB 6
#define ATOM_FDIM 133
#define BOND_FDIM 147
#define KP_IN  160   // BOND_FDIM padded to mult of 32
#define KP_HID 256
#define KP_OUT 416   // (ATOM_FDIM + HIDDEN)=389 padded to mult of 32

// ---------------------------------------------------------------------------
// fp32 -> bf16 row convert with K padding (zeros in the pad region)
// ---------------------------------------------------------------------------
__global__ __launch_bounds__(256)
void convert_pad_kernel(const float* __restrict__ src, __bf16* __restrict__ dst,
                        int rows, int ksrc, int kdst) {
    size_t total = (size_t)rows * (size_t)kdst;
    for (size_t i = (size_t)blockIdx.x * 256 + threadIdx.x; i < total;
         i += (size_t)gridDim.x * 256) {
        int r = (int)(i / kdst);
        int k = (int)(i % kdst);
        float v = (k < ksrc) ? src[(size_t)r * ksrc + k] : 0.0f;
        dst[i] = (__bf16)v;
    }
}

// ---------------------------------------------------------------------------
// a_input = concat(f_atoms, a_message) in bf16, padded to KP_OUT
// ---------------------------------------------------------------------------
__global__ __launch_bounds__(256)
void concat_kernel(const float* __restrict__ f_atoms,
                   const __bf16* __restrict__ amsg,
                   __bf16* __restrict__ ainput, int rows) {
    size_t total = (size_t)rows * KP_OUT;
    for (size_t i = (size_t)blockIdx.x * 256 + threadIdx.x; i < total;
         i += (size_t)gridDim.x * 256) {
        int r = (int)(i / KP_OUT);
        int k = (int)(i % KP_OUT);
        float v;
        if (k < ATOM_FDIM)               v = f_atoms[(size_t)r * ATOM_FDIM + k];
        else if (k < ATOM_FDIM + HIDDEN) v = (float)amsg[(size_t)r * HIDDEN + (k - ATOM_FDIM)];
        else                             v = 0.0f;
        ainput[i] = (__bf16)v;
    }
}

// ---------------------------------------------------------------------------
// a_message[a] = sum_j message[a2b[a][j]]   (one wave per atom row, 16B/lane)
// ---------------------------------------------------------------------------
__global__ __launch_bounds__(256)
void gather_kernel(const __bf16* __restrict__ msg, const int* __restrict__ a2b,
                   __bf16* __restrict__ amsg, int n_rows) {
    int row  = blockIdx.x * 8 + (threadIdx.x >> 5);
    if (row >= n_rows) return;
    int lane = threadIdx.x & 31;
    const int* nb = a2b + (size_t)row * MAX_NB;
    float acc[8];
#pragma unroll
    for (int i = 0; i < 8; ++i) acc[i] = 0.0f;
#pragma unroll
    for (int j = 0; j < MAX_NB; ++j) {
        int b = nb[j];
        bf16x8 v = *(const bf16x8*)(msg + (size_t)b * HIDDEN + lane * 8);
#pragma unroll
        for (int i = 0; i < 8; ++i) acc[i] += (float)v[i];
    }
    bf16x8 o;
#pragma unroll
    for (int i = 0; i < 8; ++i) o[i] = (__bf16)acc[i];
    *(bf16x8*)(amsg + (size_t)row * HIDDEN + lane * 8) = o;
}

// ---------------------------------------------------------------------------
// msg_in[b] = a_message[b2a[b]] - message[b2revb[b]]   (one wave per bond)
// ---------------------------------------------------------------------------
__global__ __launch_bounds__(256)
void bond_kernel(const __bf16* __restrict__ amsg, const __bf16* __restrict__ msg,
                 const int* __restrict__ b2a, const int* __restrict__ b2revb,
                 __bf16* __restrict__ msg_in, int n_rows) {
    int row  = blockIdx.x * 8 + (threadIdx.x >> 5);
    if (row >= n_rows) return;
    int lane = threadIdx.x & 31;
    int a  = b2a[row];
    int rb = b2revb[row];
    bf16x8 x = *(const bf16x8*)(amsg + (size_t)a  * HIDDEN + lane * 8);
    bf16x8 y = *(const bf16x8*)(msg  + (size_t)rb * HIDDEN + lane * 8);
    bf16x8 o;
#pragma unroll
    for (int i = 0; i < 8; ++i) o[i] = (__bf16)((float)x[i] - (float)y[i]);
    *(bf16x8*)(msg_in + (size_t)row * HIDDEN + lane * 8) = o;
}

// ---------------------------------------------------------------------------
// WMMA GEMM: out[M][256] = epilogue(A[M][KP] @ B[256][KP]^T)
//   block = 256 threads = 8 waves; block tile 128 rows x 64 cols (gridDim.y=4)
//   wave tile 16 rows x 64 cols => 4 accumulator tiles, v_wmma_f32_16x16x32_bf16
//   B panel (64 cols x KP) in LDS, bank-conflict padded (LDB = KP+8).
//   KP==256: panel staged by the Tensor Data Mover (tensor_load_to_lds) with
//            pad_interval=128dw / pad_amount=4dw reproducing the LDB layout.
//   Epilogue transposes accumulators through LDS for 16B/lane global stores.
// MODE 0: preact = acc (bf16), out = relu(acc)                 [input GEMM]
// MODE 1: out = relu(preact + acc)                             [hidden GEMM]
// MODE 2: out_f32 = relu(acc + bias[n])                        [readout GEMM]
// ---------------------------------------------------------------------------
template <int MODE, int KP>
__global__ __launch_bounds__(256)
void gemm_wmma_kernel(const __bf16* __restrict__ A, const __bf16* __restrict__ B,
                      const __bf16* __restrict__ preact_in,
                      const float* __restrict__ bias,
                      __bf16* __restrict__ out_bf, __bf16* __restrict__ preact_out,
                      float* __restrict__ out_f, int M) {
    constexpr int LDB = KP + 8;                 // +16B pad to spread LDS banks
    constexpr int B_BYTES = 64 * LDB * 2;
    constexpr int SMEM_BYTES = (B_BYTES > 32768) ? B_BYTES : 32768;
    __shared__ char smem[SMEM_BYTES];
    __bf16* Bs = (__bf16*)smem;

    const int tid  = threadIdx.x;
    const int wave = tid >> 5;
    const int lane = tid & 31;
    const int nBase   = blockIdx.y * 64;
    const int rowBase = blockIdx.x * 128 + wave * 16;

    // ---- stage the 64-column B panel into LDS ----
    if (KP == 256) {
        // Tensor Data Mover: 2D tile 256(el) x 64 rows, data_size=2B,
        // LDS padding: 4 dwords after every 128 dwords -> row stride = LDB*2.
        if (tid == 0) {
            unsigned long long ga = (unsigned long long)(size_t)(B + (size_t)nBase * KP);
            unsigned int la = (unsigned int)(size_t)&Bs[0];
            u32x4 g0 = { 1u,                                   // count=1, user desc
                         la,                                   // lds_addr
                         (unsigned int)ga,                     // global_addr[31:0]
                         (unsigned int)(ga >> 32) | (2u << 30) };   // addr[56:32] | type=2
            i32x8 g1 = { (int)((1u << 16) | (1u << 20) | (6u << 22) | (3u << 25)),
                         //  data_size=2B   pad_en      pad_int=128dw  pad_amt=4dw
                         (int)(256u << 16),   // tensor_dim0 = 256 (bits 79:48)
                         (int)(64u << 16),    // tensor_dim1 = 64  (bits 111:80)
                         (int)(256u << 16),   // tile_dim0 = 256   (bits 127:112)
                         64,                  // tile_dim1 = 64, tile_dim2 = 0
                         256, 0, 0 };         // tensor_dim0_stride = 256
            i32x4 z4 = { 0, 0, 0, 0 };
#if defined(__clang_major__) && (__clang_major__ >= 23)
            i32x8 z8 = { 0, 0, 0, 0, 0, 0, 0, 0 };
            __builtin_amdgcn_tensor_load_to_lds(g0, g1, z4, z4, z8, 0);
#else
            __builtin_amdgcn_tensor_load_to_lds(g0, g1, z4, z4, 0);
#endif
            __builtin_amdgcn_s_wait_tensorcnt(0);
        }
    } else {
        constexpr int chunksPerRow = KP / 8;    // 16B chunks
        constexpr int totalChunks  = 64 * chunksPerRow;
        for (int c = tid; c < totalChunks; c += 256) {
            int n = c / chunksPerRow;
            int k = (c % chunksPerRow) * 8;
            *(bf16x8*)&Bs[n * LDB + k] = *(const bf16x8*)&B[(size_t)(nBase + n) * KP + k];
        }
    }
    __syncthreads();

    const int halfSel = (lane >> 4) * 8;        // K sub-block select per ISA layout
    const int mLocal  = lane & 15;
    int rowA = rowBase + mLocal;
    if (rowA >= M) rowA = M - 1;                // clamp; stores are guarded below
    const __bf16* arow = A + (size_t)rowA * KP;

    floatx8 acc[4] = {};
    for (int k0 = 0; k0 < KP; k0 += 32) {
        if (k0 + 128 < KP) __builtin_prefetch(arow + k0 + 128, 0, 3);
        // A operand: row mLocal, K {k0+h..+7, k0+16+h..+23}, h = halfSel
        bf16x8 alo = *(const bf16x8*)(arow + k0 + halfSel);
        bf16x8 ahi = *(const bf16x8*)(arow + k0 + 16 + halfSel);
        bf16x16 a;
#pragma unroll
        for (int i = 0; i < 8; ++i) { a[i] = alo[i]; a[i + 8] = ahi[i]; }

        // manually unrolled so the reuse_a immediate is a literal constant;
        // A operand is identical across the 4 tiles -> reuse_a on t>0.
#define WMMA_TILE(T, RA)                                                      \
        {                                                                     \
            const __bf16* brow = &Bs[((T) * 16 + mLocal) * LDB + k0 + halfSel]; \
            bf16x8 blo = *(const bf16x8*)(brow);                              \
            bf16x8 bhi = *(const bf16x8*)(brow + 16);                         \
            bf16x16 b;                                                        \
            _Pragma("unroll")                                                 \
            for (int i = 0; i < 8; ++i) { b[i] = blo[i]; b[i + 8] = bhi[i]; } \
            acc[T] = __builtin_amdgcn_wmma_f32_16x16x32_bf16(                 \
                false, a, false, b, (short)0, acc[T], RA, false);             \
        }
        WMMA_TILE(0, false)
        WMMA_TILE(1, true)
        WMMA_TILE(2, true)
        WMMA_TILE(3, true)
#undef WMMA_TILE
    }

    // ---- epilogue: transpose accumulators through LDS for coalesced I/O ----
    // D layout: element (m = r + 8*(lane>=16), n = lane&15) in acc[t][r].
    __syncthreads();                            // B panel no longer needed
    float* wstage = (float*)smem + wave * 1024; // 16 rows x 64 cols fp32
    const int mOff = (lane >> 4) * 8;
#pragma unroll
    for (int t = 0; t < 4; ++t)
#pragma unroll
        for (int r = 0; r < 8; ++r)
            wstage[(mOff + r) * 64 + t * 16 + (lane & 15)] = acc[t][r];

#pragma unroll
    for (int c = 0; c < 8; ++c) {
        int e   = (c * 32 + lane) * 4;          // fp32 element index in wave tile
        int r   = e >> 6;                       // local row
        int cc  = e & 63;                       // local col
        int row = rowBase + r;
        if (row >= M) continue;
        int col = nBase + cc;
        floatx4 v = *(floatx4*)&wstage[e];      // contiguous 16B ds load
        size_t idx = (size_t)row * HIDDEN + col;
        if (MODE == 0) {
            bf16x4 p, q;
#pragma unroll
            for (int i = 0; i < 4; ++i) {
                p[i] = (__bf16)v[i];
                q[i] = (__bf16)fmaxf(v[i], 0.0f);
            }
            *(bf16x4*)&preact_out[idx] = p;
            *(bf16x4*)&out_bf[idx]     = q;
        } else if (MODE == 1) {
            bf16x4 pe = *(const bf16x4*)&preact_in[idx];
            bf16x4 q;
#pragma unroll
            for (int i = 0; i < 4; ++i)
                q[i] = (__bf16)fmaxf((float)pe[i] + v[i], 0.0f);
            *(bf16x4*)&out_bf[idx] = q;
        } else {
            floatx4 bb = *(const floatx4*)&bias[col];
            floatx4 o;
#pragma unroll
            for (int i = 0; i < 4; ++i) o[i] = fmaxf(v[i] + bb[i], 0.0f);
            *(floatx4*)&out_f[idx] = o;
        }
    }
}

// ---------------------------------------------------------------------------
// per-molecule mean over 30 contiguous atom rows (skip pad row 0)
// ---------------------------------------------------------------------------
__global__ __launch_bounds__(256)
void mol_mean_kernel(const float* __restrict__ ah, float* __restrict__ out) {
    int mol = blockIdx.x;
    int h   = threadIdx.x;
    size_t base = ((size_t)mol * ATOMS_PER_MOL + 1) * HIDDEN + h;
    float s = 0.0f;
#pragma unroll 6
    for (int i = 0; i < ATOMS_PER_MOL; ++i) s += ah[base + (size_t)i * HIDDEN];
    out[(size_t)mol * HIDDEN + h] = s * (1.0f / ATOMS_PER_MOL);
}

// ---------------------------------------------------------------------------
static inline size_t align256(size_t x) { return (x + 255) & ~(size_t)255; }

extern "C" void kernel_launch(void* const* d_in, const int* in_sizes, int n_in,
                              void* d_out, int out_size, void* d_ws, size_t ws_size,
                              hipStream_t stream) {
    const float* f_atoms = (const float*)d_in[0];   // [A+1,133]
    const float* f_bonds = (const float*)d_in[1];   // [B+1,147]
    const int*   a2b     = (const int*)d_in[2];     // [A+1,6]
    const int*   b2a     = (const int*)d_in[3];     // [B+1]
    const int*   b2revb  = (const int*)d_in[4];     // [B+1]
    const float* W_i     = (const float*)d_in[6];   // [256,147]
    const float* W_h     = (const float*)d_in[7];   // [256,256]
    const float* W_o     = (const float*)d_in[8];   // [256,389]
    const float* b_o     = (const float*)d_in[9];   // [256]
    float* out = (float*)d_out;                     // [5000,256]

    const int nB = N_BONDS + 1;                     // 320001 bond rows
    const int nA = N_ATOMS + 1;                     // 150001 atom rows

    // ---- workspace carve-up (bf16 activations to halve HBM traffic) ----
    char* ws = (char*)d_ws;
    size_t off = 0;
    __bf16* fbonds_bf = (__bf16*)(ws + off); off += align256((size_t)nB * KP_IN  * 2);
    __bf16* Wi_bf     = (__bf16*)(ws + off); off += align256((size_t)256 * KP_IN  * 2);
    __bf16* Wh_bf     = (__bf16*)(ws + off); off += align256((size_t)256 * KP_HID * 2);
    __bf16* Wo_bf     = (__bf16*)(ws + off); off += align256((size_t)256 * KP_OUT * 2);
    __bf16* inp_bf    = (__bf16*)(ws + off); off += align256((size_t)nB * HIDDEN * 2);
    __bf16* msg_bf    = (__bf16*)(ws + off); off += align256((size_t)nB * HIDDEN * 2);
    __bf16* msgin_bf  = (__bf16*)(ws + off); off += align256((size_t)nB * HIDDEN * 2);
    __bf16* amsg_bf   = (__bf16*)(ws + off); off += align256((size_t)nA * HIDDEN * 2);
    __bf16* ainput_bf = (__bf16*)(ws + off); off += align256((size_t)nA * KP_OUT * 2);
    float*  ah_f32    = (float*) (ws + off); off += align256((size_t)nA * HIDDEN * 4);
    (void)ws_size; (void)in_sizes; (void)n_in; (void)out_size;

    // ---- 1. precision conversion / K padding ----
    convert_pad_kernel<<<4096, 256, 0, stream>>>(f_bonds, fbonds_bf, nB, BOND_FDIM, KP_IN);
    convert_pad_kernel<<<161,  256, 0, stream>>>(W_i, Wi_bf, 256, BOND_FDIM, KP_IN);
    convert_pad_kernel<<<256,  256, 0, stream>>>(W_h, Wh_bf, 256, HIDDEN, KP_HID);
    convert_pad_kernel<<<417,  256, 0, stream>>>(W_o, Wo_bf, 256, ATOM_FDIM + HIDDEN, KP_OUT);

    // ---- 2. input GEMM: inp = f_bonds @ W_i^T ; message = relu(inp) ----
    {
        dim3 grid((nB + 127) / 128, HIDDEN / 64);
        gemm_wmma_kernel<0, KP_IN><<<grid, 256, 0, stream>>>(
            fbonds_bf, Wi_bf, nullptr, nullptr, msg_bf, inp_bf, nullptr, nB);
    }

    // ---- 3. message passing: DEPTH-1 = 5 iterations ----
    const int atomBlocks = (nA + 7) / 8;
    const int bondBlocks = (nB + 7) / 8;
    dim3 gemmGrid((nB + 127) / 128, HIDDEN / 64);
    for (int d = 0; d < 5; ++d) {
        gather_kernel<<<atomBlocks, 256, 0, stream>>>(msg_bf, a2b, amsg_bf, nA);
        bond_kernel<<<bondBlocks, 256, 0, stream>>>(amsg_bf, msg_bf, b2a, b2revb, msgin_bf, nB);
        gemm_wmma_kernel<1, KP_HID><<<gemmGrid, 256, 0, stream>>>(
            msgin_bf, Wh_bf, inp_bf, nullptr, msg_bf, nullptr, nullptr, nB);
    }

    // ---- 4. readout ----
    gather_kernel<<<atomBlocks, 256, 0, stream>>>(msg_bf, a2b, amsg_bf, nA);
    concat_kernel<<<4096, 256, 0, stream>>>(f_atoms, amsg_bf, ainput_bf, nA);
    {
        dim3 grid((nA + 127) / 128, HIDDEN / 64);
        gemm_wmma_kernel<2, KP_OUT><<<grid, 256, 0, stream>>>(
            ainput_bf, Wo_bf, nullptr, b_o, nullptr, nullptr, ah_f32, nA);
    }

    // ---- 5. per-molecule mean ----
    mol_mean_kernel<<<N_MOLS, 256, 0, stream>>>(ah_f32, out);
}
